// APPNP_Net_57191784513887
// MI455X (gfx1250) — compile-verified, matrix operand order checked by reading, not run on hardware
//
#include <hip/hip_runtime.h>
#include <hip/hip_bf16.h>
#include <stdint.h>

// ---------------------------------------------------------------------------
// Types for gfx1250 WMMA (wave32): D(16x16 f32) = A(16x32 bf16) x B(32x16 bf16) + C
// ---------------------------------------------------------------------------
typedef __attribute__((ext_vector_type(16))) __bf16 v16bf;
typedef __attribute__((ext_vector_type(8)))  __bf16 v8bf;
typedef __attribute__((ext_vector_type(8)))  float  v8f;
typedef __attribute__((ext_vector_type(4)))  float  f32x4;

// Assemble a 16-element bf16 WMMA fragment from LDS per the CDNA5 16-bit
// A/B VGPR layout: per lane, VGPR0-3 hold 8 consecutive K (sel. by lane-half),
// VGPR4-7 hold the other 8 at +16 elements.
__device__ __forceinline__ v16bf ldsfrag(const __bf16* p) {
    v8bf lo = *(const v8bf*)(p);        // K = lh*8 .. lh*8+7
    v8bf hi = *(const v8bf*)(p + 16);   // K = 16+lh*8 .. 16+lh*8+7
    return __builtin_shufflevector(lo, hi, 0,1,2,3,4,5,6,7,8,9,10,11,12,13,14,15);
}

// ---------------------------------------------------------------------------
// Tiled bf16 WMMA GEMM:  D[M,Nt] = act(A[M,Kt] * B[Kt,Nt] + bias)
// Block tile 128x64, 8 waves (4 along M x 2 along N), wave tile 32x32.
// K staged in LDS chunks of 32 with padded stride 40 (bank-conflict break).
// Software-pipelined: next chunk's global loads overlap current chunk's WMMAs.
// Tail rows (>= M) compute on clamped row M-1 data; GEMM rows are independent
// and those accumulator rows are never stored, so no zero-fill is needed.
// ---------------------------------------------------------------------------
#define BM   128
#define BN   64
#define BK   32
#define LDP  40

__global__ __launch_bounds__(256) void gemm_bf16_kernel(
    const float* __restrict__ A, const float* __restrict__ B,
    const float* __restrict__ bias, float* __restrict__ D,
    int M, int Nt, int Kt, int do_relu)
{
    __shared__ __align__(16) unsigned short sA[BM * LDP];
    __shared__ __align__(16) unsigned short sB[BN * LDP];

    const int tid  = threadIdx.x;
    const int wid  = tid >> 5;
    const int lane = tid & 31;
    const int lm   = lane & 15;     // row/col within 16x16 tile
    const int lh   = lane >> 4;     // lane half -> K octet select

    const int m_base = blockIdx.x * BM;
    const int n_base = blockIdx.y * BN;
    const int wm = (wid & 3) << 5;  // wave row offset in block tile
    const int wn = (wid >> 2) << 5; // wave col offset in block tile

    const v8f vzero = {0.f,0.f,0.f,0.f,0.f,0.f,0.f,0.f};
    v8f acc[2][2];
    acc[0][0] = vzero; acc[0][1] = vzero; acc[1][0] = vzero; acc[1][1] = vzero;

    // cooperative-load indices
    const int arow = tid >> 1;            // 0..127
    const int akq  = (tid & 1) * 16;      // 0 or 16
    const int bcol = tid & 63;            // 0..63
    const int bkq  = (tid >> 6) * 8;      // 0,8,16,24

    const int  grow = m_base + arow;
    const int  crow = grow < M ? grow : (M - 1);      // clamped tail row
    const float* arowp = A + (size_t)crow * Kt + akq;
    const float* bcolp = B + (size_t)bkq * Nt + n_base + bcol;

    float a_reg[16];
    float b_reg[8];

    // ---- prologue: prefetch chunk 0 into registers ----
    {
        const f32x4* ap = (const f32x4*)(arowp);
        #pragma unroll
        for (int i = 0; i < 4; ++i) {
            f32x4 v = ap[i];
            a_reg[4*i+0] = v[0]; a_reg[4*i+1] = v[1];
            a_reg[4*i+2] = v[2]; a_reg[4*i+3] = v[3];
        }
        #pragma unroll
        for (int i = 0; i < 8; ++i) b_reg[i] = bcolp[(size_t)i * Nt];
    }

    for (int k0 = 0; k0 < Kt; k0 += BK) {
        // ---- store staged registers to LDS (native packed f32->bf16 cvt) ----
        {
            __bf16* sp = (__bf16*)sA + arow * LDP + akq;
            #pragma unroll
            for (int i = 0; i < 16; ++i) sp[i] = (__bf16)a_reg[i];
            __bf16* spb = (__bf16*)sB + bcol * LDP + bkq;
            #pragma unroll
            for (int i = 0; i < 8; ++i) spb[i] = (__bf16)b_reg[i];
        }
        __syncthreads();

        // ---- prefetch next chunk (overlaps WMMA below) ----
        if (k0 + BK < Kt) {
            const f32x4* ap = (const f32x4*)(arowp + k0 + BK);
            #pragma unroll
            for (int i = 0; i < 4; ++i) {
                f32x4 v = ap[i];
                a_reg[4*i+0] = v[0]; a_reg[4*i+1] = v[1];
                a_reg[4*i+2] = v[2]; a_reg[4*i+3] = v[3];
            }
            const float* bp = bcolp + (size_t)(k0 + BK) * Nt;
            #pragma unroll
            for (int i = 0; i < 8; ++i) b_reg[i] = bp[(size_t)i * Nt];
        }

        // ---- compute: 4 WMMAs on the 32x32 wave tile ----
        const __bf16* sAb = (const __bf16*)sA;
        const __bf16* sBb = (const __bf16*)sB;
        v16bf a0 = ldsfrag(sAb + (wm +  0 + lm) * LDP + lh * 8);
        v16bf a1 = ldsfrag(sAb + (wm + 16 + lm) * LDP + lh * 8);
        v16bf b0 = ldsfrag(sBb + (wn +  0 + lm) * LDP + lh * 8);
        v16bf b1 = ldsfrag(sBb + (wn + 16 + lm) * LDP + lh * 8);

        acc[0][0] = __builtin_amdgcn_wmma_f32_16x16x32_bf16(false, a0, false, b0, (short)0, acc[0][0], false, false);
        acc[0][1] = __builtin_amdgcn_wmma_f32_16x16x32_bf16(false, a0, false, b1, (short)0, acc[0][1], false, false);
        acc[1][0] = __builtin_amdgcn_wmma_f32_16x16x32_bf16(false, a1, false, b0, (short)0, acc[1][0], false, false);
        acc[1][1] = __builtin_amdgcn_wmma_f32_16x16x32_bf16(false, a1, false, b1, (short)0, acc[1][1], false, false);

        __syncthreads();
    }

    // ---- epilogue: bias (+ReLU), f32 store. C/D layout: VGPR r -> row r+8*lh, col lm ----
    #pragma unroll
    for (int ti = 0; ti < 2; ++ti) {
        #pragma unroll
        for (int tj = 0; tj < 2; ++tj) {
            int col = n_base + wn + tj * 16 + lm;
            float bv = bias[col];
            #pragma unroll
            for (int r = 0; r < 8; ++r) {
                int row = m_base + wm + ti * 16 + r + lh * 8;
                if (row < M) {
                    float v = acc[ti][tj][r] + bv;
                    if (do_relu) v = fmaxf(v, 0.0f);
                    D[(size_t)row * Nt + col] = v;
                }
            }
        }
    }
}

// ---------------------------------------------------------------------------
// Graph propagation kernels (C == 64 classes, 2 floats per lane, wave32)
// ---------------------------------------------------------------------------
__global__ __launch_bounds__(256) void init_kernel(float* __restrict__ agg,
                                                   float* __restrict__ deg,
                                                   long long total, int N)
{
    long long i = (long long)blockIdx.x * blockDim.x + threadIdx.x;
    if (i < total) agg[i] = 0.0f;
    if (i < N)     deg[i] = 1.0f;   // self-loop contributes 1 to in-degree
}

__global__ __launch_bounds__(256) void degree_kernel(const int* __restrict__ dst,
                                                     float* __restrict__ deg, int E)
{
    int i = blockIdx.x * blockDim.x + threadIdx.x;
    if (i < E) atomicAdd(&deg[__builtin_nontemporal_load(&dst[i])], 1.0f);
}

__global__ __launch_bounds__(256) void rsqrt_kernel(const float* __restrict__ deg,
                                                    float* __restrict__ nrm, int N)
{
    int i = blockIdx.x * blockDim.x + threadIdx.x;
    if (i < N) nrm[i] = rsqrtf(deg[i]);
}

// One wave per edge: gather h[src]*norm[src] (256B coalesced), scatter-add to agg[dst].
// Edge lists are read-once-per-iteration streams -> non-temporal loads so the
// resident h/agg/h0 working set (~77MB, fits 192MB L2) is not evicted.
__global__ __launch_bounds__(256) void scatter_kernel(
    const int* __restrict__ src, const int* __restrict__ dst,
    const float* __restrict__ h, const float* __restrict__ nrm,
    float* __restrict__ agg, int E)
{
    const int lane = threadIdx.x & 31;
    int wid = (blockIdx.x * blockDim.x + threadIdx.x) >> 5;
    int nw  = (gridDim.x * blockDim.x) >> 5;
    for (int e = wid; e < E; e += nw) {
        int s = __builtin_nontemporal_load(&src[e]);
        int d = __builtin_nontemporal_load(&dst[e]);
        float ns = nrm[s];
        size_t so = (size_t)s << 6, dofs = (size_t)d << 6;
        float v0 = h[so + lane]      * ns;
        float v1 = h[so + 32 + lane] * ns;
        atomicAdd(&agg[dofs + lane],      v0);
        atomicAdd(&agg[dofs + 32 + lane], v1);
    }
}

// h = (1-a)*norm*(agg + h*norm) + a*h0 ; the "h*norm" term is the self-loop edge.
// Also re-zeroes agg for the next iteration.
__global__ __launch_bounds__(256) void combine_kernel(
    float* __restrict__ h, const float* __restrict__ h0,
    float* __restrict__ agg, const float* __restrict__ nrm, long long total)
{
    long long i = (long long)blockIdx.x * blockDim.x + threadIdx.x;
    if (i >= total) return;
    int n = (int)(i >> 6);
    float nn = nrm[n];
    h[i] = 0.9f * nn * (agg[i] + h[i] * nn) + 0.1f * h0[i];
    agg[i] = 0.0f;
}

// log_softmax over 64 classes: one wave per row, shfl_xor reductions (wave32)
__global__ __launch_bounds__(256) void logsoftmax_kernel(
    const float* __restrict__ h, float* __restrict__ out, int N)
{
    const int lane = threadIdx.x & 31;
    int row = (blockIdx.x * blockDim.x + threadIdx.x) >> 5;
    if (row >= N) return;
    size_t o = (size_t)row << 6;
    float x0 = h[o + lane], x1 = h[o + 32 + lane];
    float m = fmaxf(x0, x1);
    #pragma unroll
    for (int off = 16; off; off >>= 1) m = fmaxf(m, __shfl_xor(m, off, 32));
    float s = __expf(x0 - m) + __expf(x1 - m);
    #pragma unroll
    for (int off = 16; off; off >>= 1) s += __shfl_xor(s, off, 32);
    float ls = __logf(s);
    __builtin_nontemporal_store(x0 - m - ls, &out[o + lane]);
    __builtin_nontemporal_store(x1 - m - ls, &out[o + 32 + lane]);
}

// ---------------------------------------------------------------------------
// Host launcher
// ---------------------------------------------------------------------------
extern "C" void kernel_launch(void* const* d_in, const int* in_sizes, int n_in,
                              void* d_out, int out_size, void* d_ws, size_t ws_size,
                              hipStream_t stream)
{
    const float* feat = (const float*)d_in[0];
    const float* w1   = (const float*)d_in[1];
    const float* b1   = (const float*)d_in[2];
    const float* w2   = (const float*)d_in[3];
    const float* b2   = (const float*)d_in[4];
    const int*   src  = (const int*)d_in[5];
    const int*   dst  = (const int*)d_in[6];

    const int H    = in_sizes[2];              // 256
    const int C    = in_sizes[4];              // 64
    const int F_IN = in_sizes[1] / H;          // 512
    const int N    = in_sizes[0] / F_IN;       // 100000
    const int E    = in_sizes[5];              // 3200000
    const int K_IT = 10;

    // workspace layout (fp32)
    float* h1  = (float*)d_ws;                 // [N, H]
    float* h0  = h1  + (size_t)N * H;          // [N, C]
    float* h   = h0  + (size_t)N * C;          // [N, C]
    float* agg = h   + (size_t)N * C;          // [N, C]
    float* deg = agg + (size_t)N * C;          // [N]
    float* nrm = deg + N;                      // [N]

    const long long totalNC = (long long)N * C;
    const int gM = (N + BM - 1) / BM;

    // 1) MLP encoder: h1 = relu(feat @ w1 + b1) ; h0 = h1 @ w2 + b2
    gemm_bf16_kernel<<<dim3(gM, H / BN), 256, 0, stream>>>(feat, w1, b1, h1, N, H, F_IN, 1);
    gemm_bf16_kernel<<<dim3(gM, C / BN), 256, 0, stream>>>(h1, w2, b2, h0, N, C, H, 0);
    hipMemcpyAsync(h, h0, (size_t)N * C * sizeof(float), hipMemcpyDeviceToDevice, stream);

    // 2) degrees (with self-loop) + symmetric norm; zero agg
    int gNC = (int)((totalNC + 255) / 256);
    init_kernel<<<gNC, 256, 0, stream>>>(agg, deg, totalNC, N);
    degree_kernel<<<(E + 255) / 256, 256, 0, stream>>>(dst, deg, E);
    rsqrt_kernel<<<(N + 255) / 256, 256, 0, stream>>>(deg, nrm, N);

    // 3) APPNP propagation, K iterations
    for (int k = 0; k < K_IT; ++k) {
        scatter_kernel<<<4096, 256, 0, stream>>>(src, dst, h, nrm, agg, E);
        combine_kernel<<<gNC, 256, 0, stream>>>(h, h0, agg, nrm, totalNC);
    }

    // 4) log_softmax -> output
    logsoftmax_kernel<<<(N * 32 + 255) / 256, 256, 0, stream>>>(h, (float*)d_out, N);
}
